// ManualLSTM_64106681860392
// MI455X (gfx1250) — compile-verified, matrix operand order checked by reading, not run on hardware
//
#include <hip/hip_runtime.h>
#include <hip/hip_bf16.h>

// ---------------------------------------------------------------------------
// LSTM forward for B=64, T=2048, V=32000, H=256 on gfx1250 (MI455X).
//  K1: zero h double-buffer + global barrier counter  (graph-replay safe)
//  K2: convert Wx/Wh f32 -> bf16 (B-fragment friendly row-major [g,k])
//  K3: gate_x = bf16WMMA( embed[x], Wx^T ) + bx   (fused gather+convert)
//  K4: persistent 16-WG recurrence: gates = gx_t + h @ Wh^T via resident
//      B-fragments, elementwise LSTM cell, bf16 h exchange + atomic barrier
// ---------------------------------------------------------------------------

typedef __attribute__((ext_vector_type(16))) __bf16 v16bf;
typedef __attribute__((ext_vector_type(8)))  float  v8f;

constexpr int  Bb  = 64;
constexpr int  Tt  = 2048;
constexpr int  Hd  = 256;
constexpr int  G4  = 1024;            // 4*H gate rows
constexpr int  NWG = 16;              // persistent workgroups in recurrence

// Workspace layout (bytes)
constexpr size_t GX_BYTES  = (size_t)Bb * Tt * G4 * 4;   // 536,870,912
constexpr size_t WBF_BYTES = (size_t)G4 * Hd * 2;        // 524,288
constexpr size_t H_BYTES   = (size_t)2 * Bb * Hd * 2;    // 65,536 (double buffer)

struct Bits32 { uint4 lo, hi; };

// Load a 16-element bf16 WMMA fragment (A or B) from row-major [*, K] memory.
// ISA 7.12.2: per lane, elems 0..7 = K koff..koff+7, elems 8..15 = K koff+16..+23.
// Caller passes p already offset by (row*K + kbase + koff); chunks are 16B apart +32B.
static __device__ __forceinline__ v16bf ld_frag_bf16(const __bf16* p) {
    Bits32 b;
    b.lo = *(const uint4*)(p);
    b.hi = *(const uint4*)(p + 16);
    return __builtin_bit_cast(v16bf, b);
}

// Same fragment but converting from f32 source (embedding rows).
static __device__ __forceinline__ v16bf cvt_frag_f32(const float* p) {
    float4 a0 = *(const float4*)(p);
    float4 a1 = *(const float4*)(p + 4);
    float4 a2 = *(const float4*)(p + 16);
    float4 a3 = *(const float4*)(p + 20);
    v16bf v;
    v[0]  = (__bf16)a0.x; v[1]  = (__bf16)a0.y; v[2]  = (__bf16)a0.z; v[3]  = (__bf16)a0.w;
    v[4]  = (__bf16)a1.x; v[5]  = (__bf16)a1.y; v[6]  = (__bf16)a1.z; v[7]  = (__bf16)a1.w;
    v[8]  = (__bf16)a2.x; v[9]  = (__bf16)a2.y; v[10] = (__bf16)a2.z; v[11] = (__bf16)a2.w;
    v[12] = (__bf16)a3.x; v[13] = (__bf16)a3.y; v[14] = (__bf16)a3.z; v[15] = (__bf16)a3.w;
    return v;
}

static __device__ __forceinline__ float sigmoidf_(float x) {
    return 1.0f / (1.0f + __expf(-x));
}

// ---------------------------------------------------------------------------
// K1: init h buffers + barrier counter
// ---------------------------------------------------------------------------
__global__ void __launch_bounds__(256) k_init(__bf16* hbuf, unsigned* cnt) {
    int i = blockIdx.x * 256 + threadIdx.x;
    if (i < 2 * Bb * Hd) hbuf[i] = (__bf16)0.0f;
    if (i == 0) *cnt = 0u;
}

// ---------------------------------------------------------------------------
// K2: f32 -> bf16 weight conversion
// ---------------------------------------------------------------------------
__global__ void __launch_bounds__(256) k_cvt(const float* __restrict__ s,
                                             __bf16* __restrict__ d, int n) {
    int i = blockIdx.x * 256 + threadIdx.x;
    if (i < n) d[i] = (__bf16)s[i];
}

// ---------------------------------------------------------------------------
// K3: gate_x[row, g] = sum_k embed[x[row], k] * Wx[g, k] + bx[g]
// rows = b*T + t (131072). Each wave: one 16-row tile x four 16-gate tiles.
// Block = 256 thr (8 waves). Grid = (8192/8, 1024/64) = (1024, 16).
// ---------------------------------------------------------------------------
__global__ void __launch_bounds__(256) k_gatex(const int* __restrict__ x,
                                               const float* __restrict__ embed,
                                               const __bf16* __restrict__ WxBf,
                                               const float* __restrict__ bx,
                                               float* __restrict__ gx) {
    const int wave    = threadIdx.x >> 5;
    const int lane    = threadIdx.x & 31;
    const int n       = lane & 15;
    const int halfsel = lane >> 4;
    const int koff    = halfsel * 8;

    const long rowBase = (long)(blockIdx.x * 8 + wave) * 16;
    const int  gBase   = blockIdx.y * 64;

    // token for this lane's A-row (lanes 16..31 mirror 0..15)
    const int token = x[rowBase + n];
    const float* arow = embed + (long)token * Hd;

    v8f acc[4];
#pragma unroll
    for (int t = 0; t < 4; ++t) {
        const float bv = bx[gBase + t * 16 + n];
#pragma unroll
        for (int r = 0; r < 8; ++r) acc[t][r] = bv;
    }

#pragma unroll
    for (int kk = 0; kk < 8; ++kk) {
        const int kb = kk * 32 + koff;
        const v16bf afrag = cvt_frag_f32(arow + kb);
#pragma unroll
        for (int t = 0; t < 4; ++t) {
            const int g = gBase + t * 16 + n;
            const v16bf bfrag = ld_frag_bf16(WxBf + (long)g * Hd + kb);
            acc[t] = __builtin_amdgcn_wmma_f32_16x16x32_bf16(
                false, afrag, false, bfrag, (short)0, acc[t], false, false);
        }
    }

    // C layout: vgpr r, lanes<16 -> M=r, lanes>=16 -> M=r+8 ; N=lane&15
#pragma unroll
    for (int t = 0; t < 4; ++t) {
        const int g = gBase + t * 16 + n;
#pragma unroll
        for (int r = 0; r < 8; ++r) {
            const long row = rowBase + r + halfsel * 8;
            gx[row * G4 + g] = acc[t][r];
        }
    }
}

// ---------------------------------------------------------------------------
// K4: persistent recurrence. 16 WGs x 512 threads (16 waves).
// WG w owns channels [w*16, w*16+16): its 64 gate rows {S, S+256, S+512, S+768}.
// Wave (bt, gt) computes gate-type gt for batch tile bt; Wh B-fragments are
// resident in VGPRs for the entire scan. h exchanged as bf16 via global
// double buffer + device-scope atomic barrier; c lives in f32 registers.
// ---------------------------------------------------------------------------
__global__ void __launch_bounds__(512, 1)
k_recur(const float* __restrict__ gx, const __bf16* __restrict__ WhBf,
        __bf16* __restrict__ hbuf, unsigned* __restrict__ cnt,
        float* __restrict__ out) {
    __shared__ float lds_gates[4][4][16][16];   // [btile][gatetype][m][n] 16KB

    const int wg   = blockIdx.x;        // 0..15 -> channel slice
    const int tid  = threadIdx.x;
    const int wave = tid >> 5;
    const int lane = tid & 31;
    const int bt   = wave >> 2;         // batch tile 0..3
    const int gt   = wave & 3;          // 0=i 1=f 2=g 3=o
    const int n       = lane & 15;
    const int halfsel = lane >> 4;
    const int koff    = halfsel * 8;
    const int g0      = wg * 16;        // channel base
    const long grow   = g0 + gt * 256 + n;   // gate row in [0,1024)

    // Resident Wh B-fragments: K = 256 -> 8 fragments (64 VGPRs/lane)
    v16bf bfrag[8];
#pragma unroll
    for (int kk = 0; kk < 8; ++kk)
        bfrag[kk] = ld_frag_bf16(WhBf + grow * Hd + kk * 32 + koff);

    // Elementwise ownership: pair p -> (b = p>>4, ch = p&15); this thread owns
    // p = tid and p = tid+512. Cell state stays in f32 registers.
    const int ew_ch = tid & 15;
    const int ew_b0 = tid >> 4;         // 0..31
    const int ew_b1 = 32 + (tid >> 4);  // 32..63
    float c0 = 0.0f, c1 = 0.0f;

    for (int t = 0; t < Tt; ++t) {
        const __bf16* hcur  = hbuf + (size_t)(t & 1) * Bb * Hd;
        __bf16*       hnext = hbuf + (size_t)((t + 1) & 1) * Bb * Hd;

        // Seed accumulator with precomputed input-gate contribution gx_t
        v8f acc;
#pragma unroll
        for (int r = 0; r < 8; ++r) {
            const long b = bt * 16 + r + halfsel * 8;
            acc[r] = gx[(b * Tt + t) * G4 + grow];
        }
        // Prefetch next step's gx (global_prefetch_b8)
        if (t + 1 < Tt) {
            const long bpf = bt * 16 + n;
            __builtin_prefetch(&gx[(bpf * Tt + (t + 1)) * G4 + grow], 0, 1);
        }

        // gates += h @ Wh^T : 8 bf16 WMMAs against resident B-fragments
        const __bf16* arow = hcur + (long)(bt * 16 + n) * Hd;
#pragma unroll
        for (int kk = 0; kk < 8; ++kk) {
            const v16bf afrag = ld_frag_bf16(arow + kk * 32 + koff);
            acc = __builtin_amdgcn_wmma_f32_16x16x32_bf16(
                false, afrag, false, bfrag[kk], (short)0, acc, false, false);
        }

        // Park the gate tile in LDS for cross-wave i/f/g/o combination
#pragma unroll
        for (int r = 0; r < 8; ++r)
            lds_gates[bt][gt][r + halfsel * 8][n] = acc[r];
        __syncthreads();

        // Elementwise LSTM cell for this WG's channels (2 (b,ch) pairs/thread)
        {
            const int btile = ew_b0 >> 4, m = ew_b0 & 15;
            const float iv = lds_gates[btile][0][m][ew_ch];
            const float fv = lds_gates[btile][1][m][ew_ch];
            const float gv = lds_gates[btile][2][m][ew_ch];
            const float ov = lds_gates[btile][3][m][ew_ch];
            c0 = sigmoidf_(fv) * c0 + sigmoidf_(iv) * tanhf(gv);
            const float hv = sigmoidf_(ov) * tanhf(c0);
            if (t == Tt - 1) out[ew_b0 * Hd + g0 + ew_ch] = hv;
            else             hnext[ew_b0 * Hd + g0 + ew_ch] = (__bf16)hv;
        }
        {
            const int btile = ew_b1 >> 4, m = ew_b1 & 15;
            const float iv = lds_gates[btile][0][m][ew_ch];
            const float fv = lds_gates[btile][1][m][ew_ch];
            const float gv = lds_gates[btile][2][m][ew_ch];
            const float ov = lds_gates[btile][3][m][ew_ch];
            c1 = sigmoidf_(fv) * c1 + sigmoidf_(iv) * tanhf(gv);
            const float hv = sigmoidf_(ov) * tanhf(c1);
            if (t == Tt - 1) out[ew_b1 * Hd + g0 + ew_ch] = hv;
            else             hnext[ew_b1 * Hd + g0 + ew_ch] = (__bf16)hv;
        }

        // Device-wide step barrier (monotonic counter; 16 WGs are co-resident)
        if (t + 1 < Tt) {
            __threadfence();
            __syncthreads();
            if (tid == 0) {
                const unsigned target = (unsigned)NWG * (unsigned)(t + 1);
                __hip_atomic_fetch_add(cnt, 1u, __ATOMIC_RELEASE,
                                       __HIP_MEMORY_SCOPE_AGENT);
                while (__hip_atomic_load(cnt, __ATOMIC_ACQUIRE,
                                         __HIP_MEMORY_SCOPE_AGENT) < target)
                    __builtin_amdgcn_s_sleep(2);
            }
            __syncthreads();
        }
    }
}

// ---------------------------------------------------------------------------
extern "C" void kernel_launch(void* const* d_in, const int* in_sizes, int n_in,
                              void* d_out, int out_size, void* d_ws, size_t ws_size,
                              hipStream_t stream) {
    const int*   x     = (const int*)d_in[0];
    const float* embed = (const float*)d_in[1];
    const float* Wx    = (const float*)d_in[2];
    const float* bx    = (const float*)d_in[3];
    const float* Wh    = (const float*)d_in[4];
    float*       out   = (float*)d_out;

    char* ws = (char*)d_ws;
    float*    gx   = (float*)(ws);
    __bf16*   WxBf = (__bf16*)(ws + GX_BYTES);
    __bf16*   WhBf = (__bf16*)(ws + GX_BYTES + WBF_BYTES);
    __bf16*   hbuf = (__bf16*)(ws + GX_BYTES + 2 * WBF_BYTES);
    unsigned* cnt  = (unsigned*)(ws + GX_BYTES + 2 * WBF_BYTES + H_BYTES);

    // K1: zero h double-buffer + barrier counter (graph-replay deterministic)
    k_init<<<dim3(129), dim3(256), 0, stream>>>(hbuf, cnt);
    // K2: bf16 weight copies
    k_cvt<<<dim3(1024), dim3(256), 0, stream>>>(Wx, WxBf, G4 * Hd);
    k_cvt<<<dim3(1024), dim3(256), 0, stream>>>(Wh, WhBf, G4 * Hd);
    // K3: big gate_x GEMM (fused embedding gather)
    k_gatex<<<dim3(1024, 16), dim3(256), 0, stream>>>(x, embed, WxBf, bx, gx);
    // K4: persistent recurrence
    k_recur<<<dim3(NWG), dim3(512), 0, stream>>>(gx, WhBf, hbuf, cnt, out);
}
